// GlobalTUL_withSpatio_46986942218301
// MI455X (gfx1250) — matrix operand order repaired and checked by reading, not compiled, not run on hardware
//
#include <hip/hip_runtime.h>
#include <hip/hip_bf16.h>

#define HIDDEN 128

typedef __attribute__((ext_vector_type(16))) __bf16 v16bf;
typedef __attribute__((ext_vector_type(8)))  __bf16 v8bf;
typedef __attribute__((ext_vector_type(8)))  float  v8f;

static __device__ __forceinline__ v8bf cvt8(float4 a, float4 b, float sc) {
  v8bf o;
  o[0] = (__bf16)(a.x * sc); o[1] = (__bf16)(a.y * sc);
  o[2] = (__bf16)(a.z * sc); o[3] = (__bf16)(a.w * sc);
  o[4] = (__bf16)(b.x * sc); o[5] = (__bf16)(b.y * sc);
  o[6] = (__bf16)(b.z * sc); o[7] = (__bf16)(b.w * sc);
  return o;
}

// ==========================================================================
// One-shot weight converters: f32 [K][N] -> bf16 transposed [N][K].
// Layer weights are fused: Wt[n][k], k<128 root_w, 128..255 rel_w[0],
// 256..383 rel_w[1]. Pred weights padded to npad rows (zero fill) so edge
// tiles can load real zeros instead of branching.
// ==========================================================================
__global__ void build_layer_wt(const float* __restrict__ root_w,
                               const float* __restrict__ rel_w,
                               __bf16* __restrict__ wt) {
  int idx = blockIdx.x * blockDim.x + threadIdx.x;     // n*384 + k
  if (idx >= HIDDEN * 384) return;
  int n = idx / 384, k = idx % 384;
  float v;
  if (k < 128)      v = root_w[k * HIDDEN + n];
  else if (k < 256) v = rel_w[(k - 128) * HIDDEN + n];
  else              v = rel_w[HIDDEN * HIDDEN + (k - 256) * HIDDEN + n];
  wt[idx] = (__bf16)v;
}

__global__ void build_pred_wt(const float* __restrict__ pw,
                              __bf16* __restrict__ wt, int nusers, int npad) {
  int idx = blockIdx.x * blockDim.x + threadIdx.x;     // n*128 + k
  if (idx >= npad * HIDDEN) return;
  int n = idx / HIDDEN, k = idx % HIDDEN;
  wt[idx] = (__bf16)((n < nusers) ? pw[k * nusers + n] : 0.0f);
}

// ==========================================================================
// Edge-parallel scatter: one wave per edge, lane handles 4 features.
// Atomic targets (s buffers, ~102 MB) stay L2-resident (192 MB L2).
// ==========================================================================
__global__ void scatter_edges(const float* __restrict__ x,
                              const int*   __restrict__ src,
                              const int*   __restrict__ dst,
                              const int*   __restrict__ etype,
                              float* __restrict__ s0, float* __restrict__ s1,
                              float* __restrict__ c0, float* __restrict__ c1,
                              int nedges) {
  long long gid = (long long)blockIdx.x * blockDim.x + threadIdx.x;
  int e    = (int)(gid >> 5);
  int lane = (int)(gid & 31);
  if (e >= nedges) return;
  int sn = src[e];
  int dn = dst[e];
  int r  = etype[e];
  float* sb = r ? s1 : s0;
  float* cb = r ? c1 : c0;
  const float4 v = *(const float4*)(x + (long long)sn * HIDDEN + lane * 4);
  float* o = sb + (long long)dn * HIDDEN + lane * 4;
  atomicAdd(o + 0, v.x);
  atomicAdd(o + 1, v.y);
  atomicAdd(o + 2, v.z);
  atomicAdd(o + 3, v.w);
  if (lane == 0) atomicAdd(cb + dn, 1.0f);
}

// ==========================================================================
// Fused RGCN layer as one K=384 WMMA GEMM:
//   out[i,:] = [x | s0/max(c0,1) | s1/max(c1,1)][i,:] @ Wt^T + bias
// Block = 256 thr = 8 wave32, one 16-row tile. Stage A tile once as bf16 in
// LDS (12 KB), then 12 x v_wmma_f32_16x16x32_bf16 per wave; A frags via
// ds_load_b128, B frags via global_load_b128 from pre-transposed bf16 wts.
// In-place (out==x) safe: after staging barrier nothing reads x from global.
// ==========================================================================
__global__ void rgcn_layer_wmma(const float* __restrict__ x,
                                const float* __restrict__ s0,
                                const float* __restrict__ s1,
                                const float* __restrict__ c0,
                                const float* __restrict__ c1,
                                const __bf16* __restrict__ wt,   // [128][384]
                                const float* __restrict__ bias,
                                float* __restrict__ out) {
  __shared__ __attribute__((aligned(16))) __bf16 At[16 * 384];
  __shared__ float inv[32];

  const int tid     = threadIdx.x;
  const int rowBase = blockIdx.x * 16;

  if (tid < 32) {
    const float* cb = (tid < 16) ? c0 : c1;
    inv[tid] = 1.0f / fmaxf(cb[rowBase + (tid & 15)], 1.0f);
  }
  __syncthreads();

  // ---- stage fused A tile -> LDS bf16: 16 rows x 48 chunks of 8
  for (int cidx = tid; cidx < 16 * 48; cidx += 256) {
    const int rloc = cidx / 48;
    const int kc   = (cidx % 48) * 8;
    const int row  = rowBase + rloc;
    const float* sp; float sc; int ko;
    if (kc < 128)      { sp = x;  sc = 1.0f;           ko = kc; }
    else if (kc < 256) { sp = s0; sc = inv[rloc];      ko = kc - 128; }
    else               { sp = s1; sc = inv[16 + rloc]; ko = kc - 256; }
    const float4 v0 = *(const float4*)(sp + (long long)row * HIDDEN + ko);
    const float4 v1 = *(const float4*)(sp + (long long)row * HIDDEN + ko + 4);
    *(v8bf*)&At[rloc * 384 + kc] = cvt8(v0, v1, sc);
  }
  __syncthreads();

  // ---- WMMA accumulate
  const int lane = tid & 31;
  const int wave = tid >> 5;
  const int m    = lane & 15;
  const int half = lane >> 4;
  const int kbA  = half * 8;      // A: K chunks {kbA..+7, 16+kbA..+7}
  const int kbB  = half * 8;      // B: K chunks {kbB..+7, 16+kbB..+7}
  const int n    = wave * 16 + m;

  const __bf16* wrow = wt + (long long)n * 384;
  v8f acc = {};
#pragma unroll
  for (int kg = 0; kg < 384; kg += 32) {
    v8bf alo = *(const v8bf*)&At[m * 384 + kg + kbA];
    v8bf ahi = *(const v8bf*)&At[m * 384 + kg + 16 + kbA];
    v8bf blo = *(const v8bf*)(wrow + kg + kbB);
    v8bf bhi = *(const v8bf*)(wrow + kg + 16 + kbB);
    v16bf a = __builtin_shufflevector(alo, ahi, 0,1,2,3,4,5,6,7,8,9,10,11,12,13,14,15);
    v16bf b = __builtin_shufflevector(blo, bhi, 0,1,2,3,4,5,6,7,8,9,10,11,12,13,14,15);
    acc = __builtin_amdgcn_wmma_f32_16x16x32_bf16(false, a, false, b,
                                                  (short)0, acc, false, false);
  }

  const float bn   = bias[n];
  const int   mtop = half * 8;
#pragma unroll
  for (int v = 0; v < 8; ++v)
    out[(long long)(rowBase + mtop + v) * HIDDEN + n] = acc[v] + bn;
}

// ==========================================================================
// Prediction head: out = h @ pred_w + pred_b, [N,128] x [128,NU].
// A tile staged in LDS (4 KB); B from padded transposed bf16 weights.
// ==========================================================================
__global__ void pred_wmma(const float* __restrict__ h,
                          const __bf16* __restrict__ wt,  // [npad][128]
                          const float* __restrict__ pb,
                          float* __restrict__ out, int nusers) {
  __shared__ __attribute__((aligned(16))) __bf16 At[16 * HIDDEN];

  const int tid     = threadIdx.x;
  const int rowBase = blockIdx.x * 16;

  {                                   // 16 rows x 16 chunks = 256 chunks
    const int rloc = tid / 16;
    const int kc   = (tid % 16) * 8;
    const float4 v0 = *(const float4*)(h + (long long)(rowBase + rloc) * HIDDEN + kc);
    const float4 v1 = *(const float4*)(h + (long long)(rowBase + rloc) * HIDDEN + kc + 4);
    *(v8bf*)&At[rloc * HIDDEN + kc] = cvt8(v0, v1, 1.0f);
  }
  __syncthreads();

  const int lane  = tid & 31;
  const int wave  = tid >> 5;
  const int m     = lane & 15;
  const int half  = lane >> 4;
  const int kbA   = half * 8;
  const int kbB   = half * 8;
  const int ntile = blockIdx.y * 8 + wave;
  const int n     = ntile * 16 + m;

  const __bf16* wrow = wt + (long long)n * HIDDEN;
  v8f acc = {};
#pragma unroll
  for (int kg = 0; kg < HIDDEN; kg += 32) {
    v8bf alo = *(const v8bf*)&At[m * HIDDEN + kg + kbA];
    v8bf ahi = *(const v8bf*)&At[m * HIDDEN + kg + 16 + kbA];
    v8bf blo = *(const v8bf*)(wrow + kg + kbB);
    v8bf bhi = *(const v8bf*)(wrow + kg + 16 + kbB);
    v16bf a = __builtin_shufflevector(alo, ahi, 0,1,2,3,4,5,6,7,8,9,10,11,12,13,14,15);
    v16bf b = __builtin_shufflevector(blo, bhi, 0,1,2,3,4,5,6,7,8,9,10,11,12,13,14,15);
    acc = __builtin_amdgcn_wmma_f32_16x16x32_bf16(false, a, false, b,
                                                  (short)0, acc, false, false);
  }

  if (n < nusers) {
    const float bn   = pb[n];
    const int   mtop = half * 8;
#pragma unroll
    for (int v = 0; v < 8; ++v)
      out[(long long)(rowBase + mtop + v) * (long long)nusers + n] = acc[v] + bn;
  }
}

// ==========================================================================
extern "C" void kernel_launch(void* const* d_in, const int* in_sizes, int n_in,
                              void* d_out, int out_size, void* d_ws, size_t ws_size,
                              hipStream_t stream) {
  const int*   edge_index = (const int*)d_in[0];
  const int*   edge_type  = (const int*)d_in[1];
  const float* emb        = (const float*)d_in[2];
  const float* rel_w1     = (const float*)d_in[3];
  const float* root_w1    = (const float*)d_in[4];
  const float* bias1      = (const float*)d_in[5];
  const float* rel_w2     = (const float*)d_in[6];
  const float* root_w2    = (const float*)d_in[7];
  const float* bias2      = (const float*)d_in[8];
  const float* pred_w     = (const float*)d_in[9];
  const float* pred_b     = (const float*)d_in[10];

  const int E  = in_sizes[0] / 2;
  const int N  = in_sizes[2] / HIDDEN;        // 100000 (multiple of 16)
  const int NU = in_sizes[10];                // 1000

  const int ntiles   = (NU + 15) / 16;        // 63
  const int ngroups  = (ntiles + 7) / 8;      // 8
  const int NPAD     = ngroups * 8 * 16;      // 1024 (covers all n-tile loads)

  // workspace: s0 | s1 | c0 | c1 | h | wtl1 | wtl2 | predwt
  char*  ws        = (char*)d_ws;
  size_t featBytes = (size_t)N * HIDDEN * sizeof(float);
  size_t cntBytes  = (size_t)N * sizeof(float);
  float*  s0     = (float*)(ws);
  float*  s1     = (float*)(ws + featBytes);
  float*  c0     = (float*)(ws + 2 * featBytes);
  float*  c1     = (float*)(ws + 2 * featBytes + cntBytes);
  float*  h      = (float*)(ws + 2 * featBytes + 2 * cntBytes);
  char*   wbase  = ws + 3 * featBytes + 2 * cntBytes;
  __bf16* wtl1   = (__bf16*)(wbase);
  __bf16* wtl2   = (__bf16*)(wbase + (size_t)HIDDEN * 384 * 2);
  __bf16* predwt = (__bf16*)(wbase + 2 * (size_t)HIDDEN * 384 * 2);
  size_t zeroBytes = 2 * featBytes + 2 * cntBytes;

  const int* src = edge_index;
  const int* dst = edge_index + E;

  dim3 blk(256);
  long long sthreads = (long long)E * 32;
  dim3 sgrid((unsigned)((sthreads + 255) / 256));
  dim3 lgrid((unsigned)(N / 16));
  dim3 pgrid((unsigned)(N / 16), (unsigned)ngroups);

  // ---- one-shot bf16 transposed weight builds
  build_layer_wt<<<(HIDDEN * 384 + 255) / 256, blk, 0, stream>>>(root_w1, rel_w1, wtl1);
  build_layer_wt<<<(HIDDEN * 384 + 255) / 256, blk, 0, stream>>>(root_w2, rel_w2, wtl2);
  build_pred_wt<<<(NPAD * HIDDEN + 255) / 256, blk, 0, stream>>>(pred_w, predwt, NU, NPAD);

  // ---- layer 1
  hipMemsetAsync(s0, 0, zeroBytes, stream);
  scatter_edges<<<sgrid, blk, 0, stream>>>(emb, src, dst, edge_type, s0, s1, c0, c1, E);
  rgcn_layer_wmma<<<lgrid, blk, 0, stream>>>(emb, s0, s1, c0, c1, wtl1, bias1, h);

  // ---- layer 2 (in place on h)
  hipMemsetAsync(s0, 0, zeroBytes, stream);
  scatter_edges<<<sgrid, blk, 0, stream>>>(h, src, dst, edge_type, s0, s1, c0, c1, E);
  rgcn_layer_wmma<<<lgrid, blk, 0, stream>>>(h, s0, s1, c0, c1, wtl2, bias2, h);

  // ---- prediction head
  pred_wmma<<<pgrid, blk, 0, stream>>>(h, predwt, pred_b, (float*)d_out, NU);
}
